// StridedAttention_61392262529135
// MI455X (gfx1250) — matrix operand order repaired and checked by reading, not compiled, not run on hardware
//
#include <hip/hip_runtime.h>

// ---------------------------------------------------------------------------
// Strided attention for MI455X (gfx1250), bf16 WMMA pipeline, f32 softmax.
// B=2, S=2048, D=1024, H=16, Dh=64, stride read from device memory (64).
// GEMM: double-buffered async global->LDS staging (ASYNCcnt, in-order
// completion) overlapping the next tile's copy with current WMMA compute.
// ---------------------------------------------------------------------------

typedef __attribute__((ext_vector_type(16))) __bf16 v16bf;
typedef __attribute__((ext_vector_type(8)))  float  v8f;

#define SLEN   2048
#define DMODEL 1024
#define NH     16
#define DHEAD  64
#define MTOT   4096   // B*S

union Frag { v16bf v; uint4 q[2]; };

__device__ __forceinline__ unsigned short f2bf(float f) {
    unsigned int u = __float_as_uint(f);
    u += 0x7FFFu + ((u >> 16) & 1u);
    return (unsigned short)(u >> 16);
}

__device__ __forceinline__ float fast_exp2(float x) {
    return __builtin_amdgcn_exp2f(x);      // v_exp_f32
}

// Per-lane async copy of 16 bytes global -> LDS (gfx1250, ASYNCcnt-tracked).
__device__ __forceinline__ void async_g2l_b128(unsigned ldsaddr, unsigned off,
                                               const void* base) {
    asm volatile("global_load_async_to_lds_b128 %0, %1, %2"
                 :: "v"(ldsaddr), "v"(off),
                    "s"((unsigned long long)(size_t)base)
                 : "memory");
}
// Async loads complete in order: waiting <=4 means everything except the
// 4 most recently issued copies (the next tile) has landed in LDS.
__device__ __forceinline__ void wait_async_le4() {
    asm volatile("s_wait_asynccnt 4" ::: "memory");
}
__device__ __forceinline__ void wait_async0() {
    asm volatile("s_wait_asynccnt 0" ::: "memory");
}

// ---------------------------------------------------------------------------
// Elementwise f32 -> bf16
// ---------------------------------------------------------------------------
__global__ __launch_bounds__(256) void cvt_bf16(const float* __restrict__ in,
                                                unsigned short* __restrict__ out,
                                                int n) {
    int i = blockIdx.x * 256 + threadIdx.x;
    if (i < n) out[i] = f2bf(in[i]);
}

// ---------------------------------------------------------------------------
// C[m,n] = sum_k A[m,k]*W[n,k]   (A: [4096,1024] bf16, W: [1024,1024] bf16)
// mode 0: scatter bf16 to [B,H,S,Dh]     (Q, K)
// mode 1: scatter bf16 to [B,H,Dh,S]     (V transposed)
// mode 2: f32 out[m*D+n] = acc + bias[n] (final projection)
// Block tile 128x128, 8 waves, each wave 32(M) x 64(N) = 2x4 WMMA tiles.
// ---------------------------------------------------------------------------
__global__ __launch_bounds__(256) void gemm_wmma(
        const unsigned short* __restrict__ A,
        const unsigned short* __restrict__ W,
        unsigned short* __restrict__ out_bf,
        float* __restrict__ out_f,
        const float* __restrict__ bias,
        int mode) {
    __shared__ unsigned short Xs[2][128 * 40];   // double-buffered, 32+8 pad
    __shared__ unsigned short Ws[2][128 * 40];

    const int tid  = threadIdx.x;
    const int lane = tid & 31;
    const int wid  = tid >> 5;
    const int wm   = wid & 3;     // 4 waves along M (32 rows each)
    const int wn   = wid >> 2;    // 2 waves along N (64 cols each)
    const int bm   = blockIdx.x >> 3;   // 32 blocks of 128 rows
    const int bn   = blockIdx.x & 7;    // 8 blocks of 128 cols
    const int m0   = bm * 128, n0 = bn * 128;
    const int half = lane >> 4, li = lane & 15;

    v8f acc[2][4];
#pragma unroll
    for (int mt = 0; mt < 2; ++mt)
#pragma unroll
        for (int nt = 0; nt < 4; ++nt)
#pragma unroll
            for (int e = 0; e < 8; ++e) acc[mt][nt][e] = 0.f;

    // Issue the 4 async copies for one 128x32 A-tile + W-tile into buffer buf.
    auto stage = [&](int buf, int kk) {
#pragma unroll
        for (int it = 0; it < 2; ++it) {
            int c    = tid + it * 256;      // 512 x 16B chunks per tile
            int row  = c >> 2;
            int col8 = (c & 3) << 3;
            unsigned ldsX = (unsigned)(size_t)&Xs[buf][row * 40 + col8];
            unsigned ldsW = (unsigned)(size_t)&Ws[buf][row * 40 + col8];
            unsigned offX = (unsigned)(((m0 + row) * DMODEL + kk + col8) * 2);
            unsigned offW = (unsigned)(((n0 + row) * DMODEL + kk + col8) * 2);
            async_g2l_b128(ldsX, offX, A);
            async_g2l_b128(ldsW, offW, W);
        }
    };

    const int NIT = DMODEL / 32;            // 32 K-steps
    stage(0, 0);                            // prologue

    for (int it = 0; it < NIT; ++it) {
        const int cur = it & 1;
        if (it + 1 < NIT) {
            stage(cur ^ 1, (it + 1) * 32);  // overlap next tile's copy
            wait_async_le4();               // current tile's copies landed
        } else {
            wait_async0();
        }
        __syncthreads();                    // all waves' copies visible

        Frag a[2], b[4];
#pragma unroll
        for (int mt = 0; mt < 2; ++mt) {
            const unsigned short* r = &Xs[cur][(wm * 32 + mt * 16 + li) * 40];
            a[mt].q[0] = *(const uint4*)&r[8 * half];        // K 0-7 / 8-15
            a[mt].q[1] = *(const uint4*)&r[16 + 8 * half];   // K 16-23 / 24-31
        }
#pragma unroll
        for (int nt = 0; nt < 4; ++nt) {
            const unsigned short* r = &Ws[cur][(wn * 64 + nt * 16 + li) * 40];
            b[nt].q[0] = *(const uint4*)&r[8 * half];        // K 0-7 / 8-15
            b[nt].q[1] = *(const uint4*)&r[16 + 8 * half];   // K 16-23 / 24-31
        }
#pragma unroll
        for (int mt = 0; mt < 2; ++mt)
#pragma unroll
            for (int nt = 0; nt < 4; ++nt)
                acc[mt][nt] = __builtin_amdgcn_wmma_f32_16x16x32_bf16(
                    false, a[mt].v, false, b[nt].v, (short)0, acc[mt][nt],
                    false, false);

        __syncthreads();                    // reads done before overwrite
    }

#pragma unroll
    for (int mt = 0; mt < 2; ++mt)
#pragma unroll
        for (int nt = 0; nt < 4; ++nt)
#pragma unroll
            for (int e = 0; e < 8; ++e) {
                int m = m0 + wm * 32 + mt * 16 + e + 8 * half;
                int n = n0 + wn * 64 + nt * 16 + li;
                float vv = acc[mt][nt][e];
                if (mode == 2) {
                    out_f[(size_t)m * DMODEL + n] = vv + bias[n];
                } else {
                    int bb = m >> 11, s = m & (SLEN - 1);
                    int h = n >> 6, dh = n & 63;
                    if (mode == 0)
                        out_bf[((((size_t)bb * NH + h) * SLEN + s) << 6) + dh] = f2bf(vv);
                    else
                        out_bf[(((size_t)bb * NH + h) * DHEAD + dh) * SLEN + s] = f2bf(vv);
                }
            }
}

// ---------------------------------------------------------------------------
// Strided flash attention. One wave per 16-row query tile, 4 waves/block.
// Key tiles: 32 strided tiles (j0 = i0%stride + stride*t) + 2 band tiles
// (i0-16, i0+16), processed in pairs so PV WMMA gets K=32. Online softmax
// in f32 (exp2 domain); P transposed C->A layout via padded LDS.
// ---------------------------------------------------------------------------
__global__ __launch_bounds__(128) void attn_wmma(
        const unsigned short* __restrict__ Q,    // [B,H,S,Dh] bf16
        const unsigned short* __restrict__ K,    // [B,H,S,Dh] bf16
        const unsigned short* __restrict__ VT,   // [B,H,Dh,S] bf16
        unsigned short* __restrict__ CTX,        // [B,S,D]    bf16
        const int* __restrict__ stridep) {
    __shared__ unsigned short Pb[4][16 * 40];    // per-wave P scratch

    const int tid  = threadIdx.x;
    const int lane = tid & 31;
    const int w    = tid >> 5;
    const int half = lane >> 4, li = lane & 15;

    const int stride = *stridep;
    const int blocksPerBH = (SLEN / 16) / 4;     // 32
    const int bh = blockIdx.x / blocksPerBH;
    const int qg = blockIdx.x % blocksPerBH;
    const int b  = bh >> 4, h = bh & 15;
    const int i0 = (qg * 4 + w) * 16;

    const unsigned short* Qg = Q  + (((size_t)b * NH + h) * SLEN) * DHEAD;
    const unsigned short* Kg = K  + (((size_t)b * NH + h) * SLEN) * DHEAD;
    const unsigned short* Vg = VT + (((size_t)b * NH + h) * DHEAD) * SLEN;

    // Q A-fragments (Dh split into two K=32 chunks), loaded once.
    Frag qf[2];
    {
        const unsigned short* r = &Qg[(size_t)(i0 + li) * DHEAD];
#pragma unroll
        for (int f = 0; f < 2; ++f) {
            qf[f].q[0] = *(const uint4*)&r[32 * f + 8 * half];
            qf[f].q[1] = *(const uint4*)&r[32 * f + 16 + 8 * half];
        }
    }

    v8f O[4];
    float mrow[8], lrow[8];
#pragma unroll
    for (int t = 0; t < 4; ++t)
#pragma unroll
        for (int e = 0; e < 8; ++e) O[t][e] = 0.f;
#pragma unroll
    for (int e = 0; e < 8; ++e) { mrow[e] = -1e30f; lrow[e] = 0.f; }

    const int nst    = SLEN / stride;            // 32
    const int npairs = (nst + 2 + 1) >> 1;       // 17
    const int c0     = i0 % stride;
    const float SC   = 0.125f * 1.44269504f;     // 1/sqrt(Dh) * log2(e)

    for (int p = 0; p < npairs; ++p) {
        int  jt[2];
        bool vld[2];
#pragma unroll
        for (int t = 0; t < 2; ++t) {
            int u = 2 * p + t;
            if (u < nst)        { jt[t] = c0 + u * stride; vld[t] = true; }
            else if (u == nst)  { jt[t] = i0 - 16; vld[t] = (i0 >= 16) && (stride > 16); }
            else                { jt[t] = i0 + 16; vld[t] = (i0 + 32 <= SLEN) && (stride > 16); }
            if (!vld[t]) jt[t] = 0;   // dummy tile, fully masked below
        }

        // Warm L0/L2 for the next pair's strided K rows (8KB-apart tiles).
        {
            int un = 2 * (p + 1);
            if (un < nst)
                __builtin_prefetch(&Kg[(size_t)(c0 + un * stride + li) * DHEAD], 0, 1);
        }

        // ---- scores for the two key tiles --------------------------------
        v8f sc[2];
#pragma unroll
        for (int t = 0; t < 2; ++t) {
            Frag kb[2];
            const unsigned short* r = &Kg[(size_t)(jt[t] + li) * DHEAD];
#pragma unroll
            for (int f = 0; f < 2; ++f) {
                kb[f].q[0] = *(const uint4*)&r[32 * f + 8 * half];
                kb[f].q[1] = *(const uint4*)&r[32 * f + 16 + 8 * half];
            }
            v8f z;
#pragma unroll
            for (int e = 0; e < 8; ++e) z[e] = 0.f;
            z = __builtin_amdgcn_wmma_f32_16x16x32_bf16(false, qf[0].v, false,
                    kb[0].v, (short)0, z, false, false);
            z = __builtin_amdgcn_wmma_f32_16x16x32_bf16(false, qf[1].v, false,
                    kb[1].v, (short)0, z, false, false);
#pragma unroll
            for (int e = 0; e < 8; ++e) {
                int i = i0 + e + 8 * half;
                int j = jt[t] + li;
                int d = i - j; d = d < 0 ? -d : d;
                bool ok = vld[t] && (d <= 1 || (d % stride) == 0);
                z[e] = ok ? z[e] * SC : -1e30f;
            }
            sc[t] = z;
        }

        // ---- online softmax update (per row, reduced across 16 lanes) ----
        float p0[8], p1[8];
#pragma unroll
        for (int e = 0; e < 8; ++e) {
            float tm = fmaxf(sc[0][e], sc[1][e]);
            tm = fmaxf(tm, __shfl_xor(tm, 1));
            tm = fmaxf(tm, __shfl_xor(tm, 2));
            tm = fmaxf(tm, __shfl_xor(tm, 4));
            tm = fmaxf(tm, __shfl_xor(tm, 8));
            float mnew  = fmaxf(mrow[e], tm);
            float alpha = fast_exp2(mrow[e] - mnew);
            mrow[e] = mnew;
#pragma unroll
            for (int t = 0; t < 4; ++t) O[t][e] *= alpha;
            p0[e] = (sc[0][e] < -1e29f) ? 0.f : fast_exp2(sc[0][e] - mnew);
            p1[e] = (sc[1][e] < -1e29f) ? 0.f : fast_exp2(sc[1][e] - mnew);
            float rs = p0[e] + p1[e];
            rs += __shfl_xor(rs, 1);
            rs += __shfl_xor(rs, 2);
            rs += __shfl_xor(rs, 4);
            rs += __shfl_xor(rs, 8);
            lrow[e] = lrow[e] * alpha + rs;
        }

        // ---- P: C-layout f32 -> A-layout bf16 via LDS --------------------
        unsigned short* pb = Pb[w];
#pragma unroll
        for (int e = 0; e < 8; ++e) {
            int row = e + 8 * half;
            pb[row * 40 + li]      = f2bf(p0[e]);
            pb[row * 40 + 16 + li] = f2bf(p1[e]);
        }
        __syncthreads();
        Frag pf;
        {
            const unsigned short* r = &pb[li * 40];
            pf.q[0] = *(const uint4*)&r[16 * half];
            pf.q[1] = *(const uint4*)&r[16 * half + 8];
        }

        // ---- O += P(16x32) * V(32x16) per Dh chunk -----------------------
#pragma unroll
        for (int t = 0; t < 4; ++t) {
            Frag vf;
            const unsigned short* r =
                &Vg[(size_t)(t * 16 + li) * SLEN + jt[half]];
            vf.q[0] = *(const uint4*)&r[0];
            vf.q[1] = *(const uint4*)&r[8];
            O[t] = __builtin_amdgcn_wmma_f32_16x16x32_bf16(false, pf.v, false,
                        vf.v, (short)0, O[t], false, false);
        }
    }

    // ---- normalize and write context [B,S,H*Dh] bf16 ---------------------
    unsigned short* cbase = CTX + ((size_t)b * SLEN) * DMODEL + (size_t)h * DHEAD;
#pragma unroll
    for (int t = 0; t < 4; ++t)
#pragma unroll
        for (int e = 0; e < 8; ++e) {
            int i  = i0 + e + 8 * half;
            int dh = t * 16 + li;
            cbase[(size_t)i * DMODEL + dh] = f2bf(O[t][e] / lrow[e]);
        }
}

// ---------------------------------------------------------------------------
extern "C" void kernel_launch(void* const* d_in, const int* in_sizes, int n_in,
                              void* d_out, int out_size, void* d_ws, size_t ws_size,
                              hipStream_t stream) {
    const float* x   = (const float*)d_in[0];
    const float* Wq  = (const float*)d_in[1];
    const float* Wk  = (const float*)d_in[2];
    const float* Wv  = (const float*)d_in[3];
    const float* Wo  = (const float*)d_in[4];
    const float* bo  = (const float*)d_in[5];
    const int* strp  = (const int*)d_in[6];

    unsigned short* ws = (unsigned short*)d_ws;
    const size_t nXD = (size_t)MTOT * DMODEL;       // 4,194,304
    const size_t nWW = (size_t)DMODEL * DMODEL;     // 1,048,576
    size_t oXB  = 0;
    size_t oWQ  = oXB + nXD;
    size_t oWK  = oWQ + nWW;
    size_t oWV  = oWK + nWW;
    size_t oWO  = oWV + nWW;
    size_t oQ   = oWO + nWW;
    size_t oK   = oQ  + nXD;
    size_t oVT  = oK  + nXD;
    size_t oCTX = oVT + nXD;
    size_t need = (oCTX + nXD) * sizeof(unsigned short);   // ~48 MB
    if (ws_size < need) return;

    // f32 -> bf16 conversions
    cvt_bf16<<<(int)((nXD + 255) / 256), 256, 0, stream>>>(x,  ws + oXB, (int)nXD);
    cvt_bf16<<<(int)((nWW + 255) / 256), 256, 0, stream>>>(Wq, ws + oWQ, (int)nWW);
    cvt_bf16<<<(int)((nWW + 255) / 256), 256, 0, stream>>>(Wk, ws + oWK, (int)nWW);
    cvt_bf16<<<(int)((nWW + 255) / 256), 256, 0, stream>>>(Wv, ws + oWV, (int)nWW);
    cvt_bf16<<<(int)((nWW + 255) / 256), 256, 0, stream>>>(Wo, ws + oWO, (int)nWW);

    // Q/K/V projections (V written transposed as [B,H,Dh,S])
    gemm_wmma<<<256, 256, 0, stream>>>(ws + oXB, ws + oWQ, ws + oQ,  nullptr, nullptr, 0);
    gemm_wmma<<<256, 256, 0, stream>>>(ws + oXB, ws + oWK, ws + oK,  nullptr, nullptr, 0);
    gemm_wmma<<<256, 256, 0, stream>>>(ws + oXB, ws + oWV, ws + oVT, nullptr, nullptr, 1);

    // strided attention -> context
    attn_wmma<<<1024, 128, 0, stream>>>(ws + oQ, ws + oK, ws + oVT, ws + oCTX, strp);

    // output projection + bias -> f32 d_out
    gemm_wmma<<<256, 256, 0, stream>>>(ws + oCTX, ws + oWO, nullptr, (float*)d_out, bo, 2);
}
